// CascadeTransformerMM_21414706938375
// MI455X (gfx1250) — 8-bit matrix operand order repaired and checked by reading, compiled, not run on hardware
//
#include <hip/hip_runtime.h>

typedef __attribute__((ext_vector_type(8))) int v8i;
typedef signed char s8;

#define Bc 8
#define Sc 1024
#define Dc 1024
#define Fc 4096
#define Lc 4
#define Mc (Bc*Sc)

__device__ __forceinline__ void async_g2l_b128(void* lds, const void* g) {
  asm volatile("global_load_async_to_lds_b128 %0, %1, off"
               :: "v"((unsigned int)(unsigned long long)lds), "v"(g)
               : "memory");
}
#define WAIT_ASYNC_0() asm volatile("s_wait_asynccnt 0x0" ::: "memory")
#define WAIT_ASYNC_3() asm volatile("s_wait_asynccnt 0x3" ::: "memory")
#define WAIT_ASYNC_5() asm volatile("s_wait_asynccnt 0x5" ::: "memory")

__device__ __forceinline__ float block_reduce_sum(float v, float* red) {
  int t = threadIdx.x;
  red[t] = v; __syncthreads();
  for (int s = 128; s > 0; s >>= 1) {
    if (t < s) red[t] += red[t + s];
    __syncthreads();
  }
  float r = red[0]; __syncthreads();
  return r;
}
__device__ __forceinline__ float block_reduce_max(float v, float* red) {
  int t = threadIdx.x;
  red[t] = v; __syncthreads();
  for (int s = 128; s > 0; s >>= 1) {
    if (t < s) red[t] = fmaxf(red[t], red[t + s]);
    __syncthreads();
  }
  float r = red[0]; __syncthreads();
  return r;
}

__global__ void __launch_bounds__(256)
wq_absmax_kernel(const float* __restrict__ Wg, const float* __restrict__ Wu,
                 const float* __restrict__ Wd, float* __restrict__ absmax) {
  __shared__ float red[256];
  const int m = blockIdx.x;
  const int l = m / 3, w = m % 3;
  const float* src = (w == 0 ? Wg : (w == 1 ? Wu : Wd)) + (size_t)l * Dc * Fc;
  const int N = Dc * Fc;
  float mx = 0.f;
  for (int i = threadIdx.x; i < N; i += 256) mx = fmaxf(mx, fabsf(src[i]));
  float r = block_reduce_max(mx, red);
  if (threadIdx.x == 0) absmax[m] = r;
}

__global__ void __launch_bounds__(256)
ternarize_kernel(const float* __restrict__ Wg, const float* __restrict__ Wu,
                 const float* __restrict__ Wd, const float* __restrict__ absmax,
                 s8* __restrict__ tw) {
  const size_t per = (size_t)Dc * Fc;
  const long long e0 = (long long)blockIdx.x * 4096;
  const int m = (int)(e0 / (long long)per);
  const int l = m / 3, w = m % 3;
  const float thr = 0.5f * (absmax[m] + 1e-5f) / 127.0f;
  const size_t base_o = (size_t)(e0 % (long long)per);
  s8* dst = tw + (size_t)w * Lc * per + (size_t)l * per;
  for (int i = 0; i < 16; ++i) {
    size_t o = base_o + (size_t)i * 256 + threadIdx.x;
    float x;
    if (w < 2) {
      size_t f = o / Dc, d = o % Dc;
      const float* src = (w == 0 ? Wg : Wu) + (size_t)l * per;
      x = src[d * Fc + f];
    } else {
      size_t d = o / Fc, f = o % Fc;
      x = Wd[(size_t)l * per + f * Dc + d];
    }
    dst[o] = (s8)(x > thr ? 1 : (x < -thr ? -1 : 0));
  }
}

__global__ void __launch_bounds__(256)
rmsnorm_quant_kernel(const float* __restrict__ x, const float* __restrict__ rs,
                     int layer, s8* __restrict__ hq, float* __restrict__ inv1) {
  __shared__ float red[256];
  const int row = blockIdx.x;
  const int t = threadIdx.x;
  const float* xr = x + (size_t)row * Dc;
  float v[4], ss = 0.f;
  for (int i = 0; i < 4; ++i) { v[i] = xr[t + 256 * i]; ss += v[i] * v[i]; }
  float sum = block_reduce_sum(ss, red);
  float rms = rsqrtf(sum * (1.0f / Dc) + 1e-6f);
  float h[4], mx = 0.f;
  for (int i = 0; i < 4; ++i) {
    h[i] = rs[layer * Dc + t + 256 * i] * (v[i] * rms);
    mx = fmaxf(mx, fabsf(h[i]));
  }
  float rmax = block_reduce_max(mx, red);
  float s = fminf(fmaxf(127.0f / (rmax + 1e-5f), 1e-3f), 1e3f);
  for (int i = 0; i < 4; ++i) {
    float q = fminf(fmaxf(rintf(s * h[i]), -128.0f), 127.0f);
    hq[(size_t)row * Dc + t + 256 * i] = (s8)(int)q;
  }
  if (t == 0) inv1[row] = 1.0f / s;
}

__device__ __forceinline__ v8i load_fragA(const s8* As, int row0, int lane) {
  const int mrow = row0 + (lane & 15);
  const int kh = (lane >> 4) * 16;
  v8i a;
  #pragma unroll
  for (int p = 0; p < 4; ++p) {
    int2 q = *(const int2*)&As[mrow * 80 + (p >> 1) * 32 + (p & 1) * 8 + kh];
    a[2 * p] = q.x; a[2 * p + 1] = q.y;
  }
  return a;
}
__device__ __forceinline__ v8i load_fragB(const s8* Bs, int col0, int lane) {
  const int ncol = col0 + (lane & 15);
  const int kh = (lane >> 4) * 16;
  int4 p0 = *(const int4*)&Bs[ncol * 80 + kh];
  int4 p1 = *(const int4*)&Bs[ncol * 80 + 32 + kh];
  v8i b = {p0.x, p0.y, p0.z, p0.w, p1.x, p1.y, p1.z, p1.w};
  return b;
}
#define WMMA_IU8(a, b, c) \
  __builtin_amdgcn_wmma_i32_16x16x64_iu8(true, (a), true, (b), (c), false, false)

__global__ void __launch_bounds__(256)
glu_gemm_kernel(const s8* __restrict__ Aq,
                const s8* __restrict__ BgT,
                const s8* __restrict__ BuT,
                const float* __restrict__ inv1,
                float* __restrict__ gout) {
  __shared__ __align__(16) s8 As[2][64 * 80];
  __shared__ __align__(16) s8 Bg[2][128 * 80];
  __shared__ __align__(16) s8 Bu[2][128 * 80];
  __shared__ float sInv[64];

  const int t = threadIdx.x, lane = t & 31, wid = t >> 5;
  const int wm = wid & 1, wn = wid >> 1;
  const int n0 = blockIdx.x * 128, m0 = blockIdx.y * 64;
  if (t < 64) sInv[t] = inv1[m0 + t];

  const int arow = t >> 2, akb = (t & 3) * 16;
  const int brow = t >> 1, bkb = (t & 1) * 32;
  const size_t aBase = (size_t)(m0 + arow) * Dc + akb;
  const size_t bBase = (size_t)(n0 + brow) * Dc + bkb;

  v8i uacc[2][2], vacc[2][2];
  #pragma unroll
  for (int i = 0; i < 2; ++i)
    #pragma unroll
    for (int j = 0; j < 2; ++j) {
      uacc[i][j] = (v8i){0, 0, 0, 0, 0, 0, 0, 0};
      vacc[i][j] = (v8i){0, 0, 0, 0, 0, 0, 0, 0};
    }

  async_g2l_b128(&As[0][arow * 80 + akb], &Aq[aBase]);
  async_g2l_b128(&Bg[0][brow * 80 + bkb], &BgT[bBase]);
  async_g2l_b128(&Bg[0][brow * 80 + bkb + 16], &BgT[bBase + 16]);
  async_g2l_b128(&Bu[0][brow * 80 + bkb], &BuT[bBase]);
  async_g2l_b128(&Bu[0][brow * 80 + bkb + 16], &BuT[bBase + 16]);

  const int KT = Dc / 64;
  for (int kt = 0; kt < KT; ++kt) {
    const int cur = kt & 1, nxt = cur ^ 1;
    if (kt + 1 < KT) {
      const size_t k1 = (size_t)(kt + 1) * 64;
      async_g2l_b128(&As[nxt][arow * 80 + akb], &Aq[aBase + k1]);
      async_g2l_b128(&Bg[nxt][brow * 80 + bkb], &BgT[bBase + k1]);
      async_g2l_b128(&Bg[nxt][brow * 80 + bkb + 16], &BgT[bBase + k1 + 16]);
      async_g2l_b128(&Bu[nxt][brow * 80 + bkb], &BuT[bBase + k1]);
      async_g2l_b128(&Bu[nxt][brow * 80 + bkb + 16], &BuT[bBase + k1 + 16]);
      WAIT_ASYNC_5();
    } else {
      WAIT_ASYNC_0();
    }
    __syncthreads();

    v8i a0 = load_fragA(As[cur], wm * 32, lane);
    v8i a1 = load_fragA(As[cur], wm * 32 + 16, lane);
    #pragma unroll
    for (int ni = 0; ni < 2; ++ni) {
      v8i bg = load_fragB(Bg[cur], wn * 32 + ni * 16, lane);
      v8i bu = load_fragB(Bu[cur], wn * 32 + ni * 16, lane);
      uacc[0][ni] = WMMA_IU8(a0, bg, uacc[0][ni]);
      uacc[1][ni] = WMMA_IU8(a1, bg, uacc[1][ni]);
      vacc[0][ni] = WMMA_IU8(a0, bu, vacc[0][ni]);
      vacc[1][ni] = WMMA_IU8(a1, bu, vacc[1][ni]);
    }
    __syncthreads();
  }

  const int kh = lane >> 4;
  #pragma unroll
  for (int mi = 0; mi < 2; ++mi)
    #pragma unroll
    for (int ni = 0; ni < 2; ++ni) {
      const int ncol = n0 + wn * 32 + ni * 16 + (lane & 15);
      #pragma unroll
      for (int r = 0; r < 8; ++r) {
        int mloc = wm * 32 + mi * 16 + kh * 8 + r;
        float is = sInv[mloc];
        float uf = (float)uacc[mi][ni][r] * is;
        float vf = (float)vacc[mi][ni][r] * is;
        float g = uf * vf / (1.0f + __expf(-uf));
        gout[(size_t)(m0 + mloc) * Fc + ncol] = g;
      }
    }
}

__global__ void __launch_bounds__(256)
actquant_g_kernel(const float* __restrict__ g, s8* __restrict__ gq,
                  float* __restrict__ inv2) {
  __shared__ float red[256];
  const int row = blockIdx.x, t = threadIdx.x;
  const float* gr = g + (size_t)row * Fc;
  float v[16], mx = 0.f;
  for (int i = 0; i < 16; ++i) { v[i] = gr[t + 256 * i]; mx = fmaxf(mx, fabsf(v[i])); }
  float rmax = block_reduce_max(mx, red);
  float s = fminf(fmaxf(127.0f / (rmax + 1e-5f), 1e-3f), 1e3f);
  for (int i = 0; i < 16; ++i) {
    float q = fminf(fmaxf(rintf(s * v[i]), -128.0f), 127.0f);
    gq[(size_t)row * Fc + t + 256 * i] = (s8)(int)q;
  }
  if (t == 0) inv2[row] = 1.0f / s;
}

__global__ void __launch_bounds__(256)
down_gemm_kernel(const s8* __restrict__ Aq,
                 const s8* __restrict__ BdT,
                 const float* __restrict__ inv2,
                 float* __restrict__ xout) {
  __shared__ __align__(16) s8 As[2][64 * 80];
  __shared__ __align__(16) s8 Bd[2][128 * 80];
  __shared__ float sInv[64];

  const int t = threadIdx.x, lane = t & 31, wid = t >> 5;
  const int wm = wid & 1, wn = wid >> 1;
  const int n0 = blockIdx.x * 128, m0 = blockIdx.y * 64;
  if (t < 64) sInv[t] = inv2[m0 + t];

  const int arow = t >> 2, akb = (t & 3) * 16;
  const int brow = t >> 1, bkb = (t & 1) * 32;
  const size_t aBase = (size_t)(m0 + arow) * Fc + akb;
  const size_t bBase = (size_t)(n0 + brow) * Fc + bkb;

  v8i acc[2][2];
  #pragma unroll
  for (int i = 0; i < 2; ++i)
    #pragma unroll
    for (int j = 0; j < 2; ++j) acc[i][j] = (v8i){0, 0, 0, 0, 0, 0, 0, 0};

  async_g2l_b128(&As[0][arow * 80 + akb], &Aq[aBase]);
  async_g2l_b128(&Bd[0][brow * 80 + bkb], &BdT[bBase]);
  async_g2l_b128(&Bd[0][brow * 80 + bkb + 16], &BdT[bBase + 16]);

  const int KT = Fc / 64;
  for (int kt = 0; kt < KT; ++kt) {
    const int cur = kt & 1, nxt = cur ^ 1;
    if (kt + 1 < KT) {
      const size_t k1 = (size_t)(kt + 1) * 64;
      async_g2l_b128(&As[nxt][arow * 80 + akb], &Aq[aBase + k1]);
      async_g2l_b128(&Bd[nxt][brow * 80 + bkb], &BdT[bBase + k1]);
      async_g2l_b128(&Bd[nxt][brow * 80 + bkb + 16], &BdT[bBase + k1 + 16]);
      WAIT_ASYNC_3();
    } else {
      WAIT_ASYNC_0();
    }
    __syncthreads();

    v8i a0 = load_fragA(As[cur], wm * 32, lane);
    v8i a1 = load_fragA(As[cur], wm * 32 + 16, lane);
    #pragma unroll
    for (int ni = 0; ni < 2; ++ni) {
      v8i b = load_fragB(Bd[cur], wn * 32 + ni * 16, lane);
      acc[0][ni] = WMMA_IU8(a0, b, acc[0][ni]);
      acc[1][ni] = WMMA_IU8(a1, b, acc[1][ni]);
    }
    __syncthreads();
  }

  const int kh = lane >> 4;
  #pragma unroll
  for (int mi = 0; mi < 2; ++mi)
    #pragma unroll
    for (int ni = 0; ni < 2; ++ni) {
      const int ncol = n0 + wn * 32 + ni * 16 + (lane & 15);
      #pragma unroll
      for (int r = 0; r < 8; ++r) {
        int mloc = wm * 32 + mi * 16 + kh * 8 + r;
        size_t o = (size_t)(m0 + mloc) * Dc + ncol;
        xout[o] = xout[o] + (float)acc[mi][ni][r] * sInv[mloc];
      }
    }
}

extern "C" void kernel_launch(void* const* d_in, const int* in_sizes, int n_in,
                              void* d_out, int out_size, void* d_ws, size_t ws_size,
                              hipStream_t stream) {
  (void)in_sizes; (void)n_in; (void)out_size; (void)ws_size;
  const float* x  = (const float*)d_in[0];
  const float* rs = (const float*)d_in[1];
  const float* Wg = (const float*)d_in[2];
  const float* Wu = (const float*)d_in[3];
  const float* Wd = (const float*)d_in[4];
  float* out = (float*)d_out;

  const size_t per = (size_t)Dc * Fc;
  char* p = (char*)d_ws;
  s8* twg = (s8*)p;                 p += (size_t)Lc * per;
  s8* twu = (s8*)p;                 p += (size_t)Lc * per;
  s8* twd = (s8*)p;                 p += (size_t)Lc * per;
  float* absmax = (float*)p;        p += 256;
  float* inv1 = (float*)p;          p += (size_t)Mc * 4;
  float* inv2 = (float*)p;          p += (size_t)Mc * 4;
  s8* hq = (s8*)p;                  p += (size_t)Mc * Dc;
  s8* gq = (s8*)p;                  p += (size_t)Mc * Fc;
  float* gbuf = (float*)p;          p += (size_t)Mc * Fc * 4;

  hipMemcpyAsync(out, x, (size_t)Mc * Dc * sizeof(float),
                 hipMemcpyDeviceToDevice, stream);

  wq_absmax_kernel<<<12, 256, 0, stream>>>(Wg, Wu, Wd, absmax);
  ternarize_kernel<<<(int)((12 * per) / 4096), 256, 0, stream>>>(Wg, Wu, Wd, absmax, twg);

  for (int l = 0; l < Lc; ++l) {
    rmsnorm_quant_kernel<<<Mc, 256, 0, stream>>>(out, rs, l, hq, inv1);
    glu_gemm_kernel<<<dim3(Fc / 128, Mc / 64), 256, 0, stream>>>(
        hq, twg + (size_t)l * per, twu + (size_t)l * per, inv1, gbuf);
    actquant_g_kernel<<<Mc, 256, 0, stream>>>(gbuf, gq, inv2);
    down_gemm_kernel<<<dim3(Dc / 128, Mc / 64), 256, 0, stream>>>(
        gq, twd + (size_t)l * per, inv2, out);
  }
}
